// MultiLevelGCN_58557584114108
// MI455X (gfx1250) — matrix operand order repaired and checked by reading, not compiled, not run on hardware
//
#include <hip/hip_runtime.h>

// ---------------------------------------------------------------------------
// Multi-level GCN for MI455X (gfx1250, wave32, WMMA bf16 16x16x32, f32 accum)
// ---------------------------------------------------------------------------

typedef __attribute__((ext_vector_type(16))) __bf16 v16bf;
typedef __attribute__((ext_vector_type(8)))  float  v8f;

// ------------------------------- utility kernels ---------------------------

__global__ void fill_zero_kernel(float* __restrict__ p, int n) {
  int i = blockIdx.x * blockDim.x + threadIdx.x;
  if (i < n) p[i] = 0.0f;
}

__global__ void deg_count_kernel(const int* __restrict__ idx,
                                 float* __restrict__ deg, int e) {
  int i = blockIdx.x * blockDim.x + threadIdx.x;
  if (i < e) atomicAdd(&deg[idx[i]], 1.0f);
}

__global__ void rsqrt_clip_kernel(float* __restrict__ d, int n) {
  int i = blockIdx.x * blockDim.x + threadIdx.x;
  if (i < n) d[i] = rsqrtf(fmaxf(d[i], 1.0f));
}

// agg[dst, :] += t[src, :], 4 features per thread; h matrices are L2-resident.
__global__ void edge_agg_kernel(const float* __restrict__ t,
                                const int* __restrict__ src,
                                const int* __restrict__ dst,
                                float* __restrict__ agg, int e, int N) {
  int per = N >> 2;
  int tid = blockIdx.x * blockDim.x + threadIdx.x;
  if (tid >= e * per) return;
  int ei = tid / per;
  int f  = (tid - ei * per) << 2;
  int s = src[ei], d = dst[ei];
  const float4 v = *(const float4*)(t + (size_t)s * N + f);
  float* o = agg + (size_t)d * N + f;
  atomicAdd(o + 0, v.x);
  atomicAdd(o + 1, v.y);
  atomicAdd(o + 2, v.z);
  atomicAdd(o + 3, v.w);
}

// out[m,n] = relu(agg[m,n] * rs[m] + bias[n]) as bf16 (feeds projection B)
__global__ void finish_bf16_kernel(const float* __restrict__ agg,
                                   const float* __restrict__ rs,
                                   const float* __restrict__ bias,
                                   __bf16* __restrict__ out, int N) {
  int m = blockIdx.y;
  int n = blockIdx.x * blockDim.x + threadIdx.x;
  if (n < N) {
    float v = agg[(size_t)m * N + n] * rs[m] + bias[n];
    out[(size_t)m * N + n] = (__bf16)fmaxf(v, 0.0f);
  }
}

// out[m,n] = agg[m,n] * rs[m] + bias[n]  (f32, final logits)
__global__ void finish_kernel(const float* __restrict__ agg,
                              const float* __restrict__ rs,
                              const float* __restrict__ bias,
                              float* __restrict__ out, int N) {
  int m = blockIdx.y;
  int n = blockIdx.x * blockDim.x + threadIdx.x;
  if (n < N) out[(size_t)m * N + n] = agg[(size_t)m * N + n] * rs[m] + bias[n];
}

// ------------------------------- WMMA GEMM ---------------------------------
// TRANSA=false: C[M,N] (+)= A[M,K] @ B[K,N]; A f32 row-major (lda=K), B f32.
// TRANSA=true : C[M,N] (+)= A^T @ B; A = P stored [K, Mtot] f32 (lda=Mtot),
//               B is bf16 (pre-converted h). Software-pipelined double-buffer:
//               issue next slab's global loads -> compute current slab from
//               LDS buffer p -> cvt/store next slab into buffer p^1 -> 1 sync.
// NT: compile-time n-tile count (straight-line WMMA chain).
// SPLITK: grid.y slices K; partials accumulated with f32 atomics.
// rowscale (optional): fused rs_out row scaling on C.

#define BPITCH 36  // halfwords; 72B row pitch -> bank-conflict-free, 8B aligned

template <bool TRANSA, int NT, bool SPLITK>
__global__ __launch_bounds__(128) void gemm_wmma_bf16(
    const float* __restrict__ A, const void* __restrict__ Bv,
    float* __restrict__ C, const float* __restrict__ rowscale,
    int M, int N, int K, int lda, int kslab) {
  constexpr bool PIPE = (TRANSA && NT == 8);
  __shared__ __bf16 Bs[PIPE ? 2 : 1][16 * NT][BPITCH];         // Bs[.][n][k]
  __shared__ __bf16 As[PIPE ? 2 : 1][TRANSA ? 64 : 2][BPITCH]; // As[.][m][k]

  const float*  Bf = (const float*)Bv;   // !TRANSA path (f32 B)
  const __bf16* Bh = (const __bf16*)Bv;  // TRANSA path (bf16 B)

  const int tid   = threadIdx.x;
  const int lane  = tid & 31;
  const int wave  = tid >> 5;
  const int mbase = blockIdx.x * 64;
  const int m0    = mbase + wave * 16;
  const int mA    = m0 + (lane & 15);
  const int mAc   = (mA < M) ? mA : (M - 1);  // clamp; OOB rows never stored
  const int kbA   = (lane >> 4) << 3;  // A frag K-offset: 0 / 8
  const int kbB   = (lane >> 4) << 4;  // unused; B frag uses kbA (0 / 8)
  const int nB    = lane & 15;

  const int k_start = SPLITK ? blockIdx.y * kslab : 0;
  int k_stop = SPLITK ? (k_start + kslab) : K;
  if (k_stop > K) k_stop = K;

  v8f acc[NT];
#pragma unroll
  for (int t = 0; t < NT; ++t) {
#pragma unroll
    for (int j = 0; j < 8; ++j) acc[t][j] = 0.0f;
  }

  if constexpr (PIPE) {
    // ================= software-pipelined projection GEMM =================
    const bool mfull = (mbase + 64 <= M);
    __bf16 bpre[32];   // column nn=tid, k=0..31 of next slab
    float4 apre[4];    // 4 float4 along m of next slab (P stream)

    auto load_slab = [&](int k0) {
      const __bf16* bp = Bh + (size_t)k0 * 128 + tid;
#pragma unroll
      for (int kk = 0; kk < 32; ++kk) bpre[kk] = bp[(size_t)kk * 128];
      if (mfull) {
#pragma unroll
        for (int u = 0; u < 4; ++u) {
          int i  = tid * 4 + u * 512;
          int kk = i >> 6;
          int mm = i & 63;
          apre[u] = *(const float4*)(A + (size_t)(k0 + kk) * lda + mbase + mm);
        }
      } else {
#pragma unroll
        for (int u = 0; u < 4; ++u) {
          int i  = tid * 4 + u * 512;
          int kk = i >> 6;
          int mm = i & 63;
          float4 v;
          v.x = (mbase + mm + 0 < M) ? A[(size_t)(k0 + kk) * lda + mbase + mm + 0] : 0.0f;
          v.y = (mbase + mm + 1 < M) ? A[(size_t)(k0 + kk) * lda + mbase + mm + 1] : 0.0f;
          v.z = (mbase + mm + 2 < M) ? A[(size_t)(k0 + kk) * lda + mbase + mm + 2] : 0.0f;
          v.w = (mbase + mm + 3 < M) ? A[(size_t)(k0 + kk) * lda + mbase + mm + 3] : 0.0f;
          apre[u] = v;
        }
      }
      // prefetch the P stream ~3 slabs ahead (global_prefetch_b8)
      int kk = ((tid * 16) >> 6) + 96;
      int mm = (tid * 16) & 63;
      if (k0 + kk < k_stop)
        __builtin_prefetch(A + (size_t)(k0 + kk) * lda + mbase + mm, 0, 1);
    };
    auto store_slab = [&](int buf) {
#pragma unroll
      for (int kk = 0; kk < 32; ++kk) Bs[buf][tid][kk] = bpre[kk];
#pragma unroll
      for (int u = 0; u < 4; ++u) {
        int i  = tid * 4 + u * 512;
        int kk = i >> 6;
        int mm = i & 63;
        As[buf][mm + 0][kk] = (__bf16)apre[u].x;
        As[buf][mm + 1][kk] = (__bf16)apre[u].y;
        As[buf][mm + 2][kk] = (__bf16)apre[u].z;
        As[buf][mm + 3][kk] = (__bf16)apre[u].w;
      }
    };

    if (k_start < k_stop) {
      load_slab(k_start);
      store_slab(0);
      __syncthreads();
      int parity = 0;
      for (int k0 = k_start; k0 < k_stop; k0 += 32) {
        const bool more = (k0 + 32) < k_stop;
        if (more) load_slab(k0 + 32);  // global loads in flight during compute
        // A fragment from LDS
        v16bf a;
        const int mloc = wave * 16 + (lane & 15);
#pragma unroll
        for (int j = 0; j < 8; ++j) {
          a[j]     = As[parity][mloc][kbA + j];
          a[j + 8] = As[parity][mloc][kbA + 16 + j];
        }
        // straight WMMA chain
#pragma unroll
        for (int nt = 0; nt < NT; ++nt) {
          v16bf b;
#pragma unroll
          for (int j = 0; j < 16; ++j) b[j] = Bs[parity][nt * 16 + nB][kbA + j + (j >> 3) * 8];
          acc[nt] = __builtin_amdgcn_wmma_f32_16x16x32_bf16(
              false, a, false, b, (short)0, acc[nt], false, false);
        }
        if (more) store_slab(parity ^ 1);
        parity ^= 1;
        __syncthreads();
      }
    }
  } else {
    // ===================== simple two-sync path (XW GEMMs) ==================
    for (int k0 = k_start; k0 < k_stop; k0 += 32) {
      __syncthreads();
      if (NT == 8) {  // N == 128 at all NT==8 call sites; thread owns column nn
        const int nn = tid;
        const float* bp = Bf + (size_t)k0 * 128 + nn;
#pragma unroll
        for (int kk = 0; kk < 32; kk += 4) {
          float f0 = bp[(size_t)(kk + 0) * 128];
          float f1 = bp[(size_t)(kk + 1) * 128];
          float f2 = bp[(size_t)(kk + 2) * 128];
          float f3 = bp[(size_t)(kk + 3) * 128];
          Bs[0][nn][kk + 0] = (__bf16)f0;  // 4 consecutive bf16 -> ds_store_b64
          Bs[0][nn][kk + 1] = (__bf16)f1;
          Bs[0][nn][kk + 2] = (__bf16)f2;
          Bs[0][nn][kk + 3] = (__bf16)f3;
        }
      } else {  // generic guarded path (tiny N=40 GEMM only)
        for (int i = tid; i < 32 * 16 * NT; i += 128) {
          int kk = i / (16 * NT);
          int nn = i - kk * (16 * NT);
          float v = (nn < N) ? Bf[(size_t)(k0 + kk) * N + nn] : 0.0f;
          Bs[0][nn][kk] = (__bf16)v;
        }
      }
      __syncthreads();

      // A fragment direct from global (row-major, clamped row)
      v16bf a;
      const float* ap = A + (size_t)mAc * lda + k0 + kbA;
      const float4 p0 = *(const float4*)(ap);
      const float4 p1 = *(const float4*)(ap + 4);
      const float4 q0 = *(const float4*)(ap + 16);
      const float4 q1 = *(const float4*)(ap + 20);
      a[0]  = (__bf16)p0.x; a[1]  = (__bf16)p0.y; a[2]  = (__bf16)p0.z; a[3]  = (__bf16)p0.w;
      a[4]  = (__bf16)p1.x; a[5]  = (__bf16)p1.y; a[6]  = (__bf16)p1.z; a[7]  = (__bf16)p1.w;
      a[8]  = (__bf16)q0.x; a[9]  = (__bf16)q0.y; a[10] = (__bf16)q0.z; a[11] = (__bf16)q0.w;
      a[12] = (__bf16)q1.x; a[13] = (__bf16)q1.y; a[14] = (__bf16)q1.z; a[15] = (__bf16)q1.w;

#pragma unroll
      for (int nt = 0; nt < NT; ++nt) {
        v16bf b;
#pragma unroll
        for (int j = 0; j < 16; ++j) b[j] = Bs[0][nt * 16 + nB][kbA + j + (j >> 3) * 8];
        acc[nt] = __builtin_amdgcn_wmma_f32_16x16x32_bf16(
            false, a, false, b, (short)0, acc[nt], false, false);
      }
    }
  }

  // ---- epilogue: C/D layout VGPR v -> row m0 + v + 8*(lane>=16), col lane&15
  const int mrow = m0 + ((lane >> 4) << 3);
#pragma unroll
  for (int nt = 0; nt < NT; ++nt) {
#pragma unroll
    for (int v = 0; v < 8; ++v) {
      int m = mrow + v;
      int n = nt * 16 + nB;
      if (m < M && n < N) {
        float val = acc[nt][v];
        if (rowscale) val *= rowscale[m];
        if (SPLITK) atomicAdd(&C[(size_t)m * N + n], val);
        else        C[(size_t)m * N + n] = val;
      }
    }
  }
}

// ------------------------------- launcher ----------------------------------

extern "C" void kernel_launch(void* const* d_in, const int* in_sizes, int n_in,
                              void* d_out, int out_size, void* d_ws,
                              size_t ws_size, hipStream_t stream) {
  const float* features = (const float*)d_in[0];
  const int*   src0 = (const int*)d_in[1];
  const int*   dst0 = (const int*)d_in[2];
  const int*   src1 = (const int*)d_in[3];
  const int*   dst1 = (const int*)d_in[4];
  const int*   src2 = (const int*)d_in[5];
  const int*   dst2 = (const int*)d_in[6];
  const float* P1 = (const float*)d_in[7];
  const float* P2 = (const float*)d_in[8];
  const float* W0 = (const float*)d_in[9];
  const float* b0 = (const float*)d_in[10];
  const float* W1 = (const float*)d_in[11];
  const float* b1 = (const float*)d_in[12];
  const float* W2 = (const float*)d_in[13];
  const float* b2 = (const float*)d_in[14];

  const int n0 = 20000, n1v = 10000, n2 = 5000;
  const int e0 = 640000, e1 = 320000, e2 = 160000;
  const int IND = 256, HID = 128, NC = 40;
  const int SPLIT = 4;

  float* out = (float*)d_out;
  float* ws  = (float*)d_ws;

  // workspace carve-up (floats)
  float* rs_out0 = ws;
  float* rs_in0  = rs_out0 + n0;
  float* rs_out1 = rs_in0 + n0;
  float* rs_in1  = rs_out1 + n1v;
  float* rs_out2 = rs_in1 + n1v;
  float* rs_in2  = rs_out2 + n2;
  float* t0   = rs_in2 + n2;                 // [n0, HID] f32
  float* agg0 = t0 + (size_t)n0 * HID;       // [n0, HID] f32 scratch
  float* h1in = agg0 + (size_t)n0 * HID;     // [n1, HID] f32
  float* t1   = h1in + (size_t)n1v * HID;    // [n1, HID] f32
  float* agg1 = t1 + (size_t)n1v * HID;      // [n1, HID] f32 scratch
  float* t2   = agg1 + (size_t)n1v * HID;    // [n2, NC]  f32
  float* agg2 = t2 + (size_t)n2 * NC;        // [n2, NC]  f32
  float* fend = agg2 + (size_t)n2 * NC;
  __bf16* h0bf = (__bf16*)fend;                              // [n0, HID] bf16
  __bf16* h1bf = (__bf16*)(fend + (size_t)n0 * HID / 2);     // [n1, HID] bf16
  float* emb  = out + (size_t)n2 * NC;       // h2in == emb, lives in d_out

  // ---- degrees: rsqrt(clip(deg,1)) for all 3 graphs in one strip ----
  const int degtot = 2 * n0 + 2 * n1v + 2 * n2;
  fill_zero_kernel<<<(degtot + 255) / 256, 256, 0, stream>>>(rs_out0, degtot);
  deg_count_kernel<<<(e0 + 255) / 256, 256, 0, stream>>>(src0, rs_out0, e0);
  deg_count_kernel<<<(e0 + 255) / 256, 256, 0, stream>>>(dst0, rs_in0, e0);
  deg_count_kernel<<<(e1 + 255) / 256, 256, 0, stream>>>(src1, rs_out1, e1);
  deg_count_kernel<<<(e1 + 255) / 256, 256, 0, stream>>>(dst1, rs_in1, e1);
  deg_count_kernel<<<(e2 + 255) / 256, 256, 0, stream>>>(src2, rs_out2, e2);
  deg_count_kernel<<<(e2 + 255) / 256, 256, 0, stream>>>(dst2, rs_in2, e2);
  rsqrt_clip_kernel<<<(degtot + 255) / 256, 256, 0, stream>>>(rs_out0, degtot);

  // ---- layer 0: t0 = (X @ W0) * rs_out0 ; scatter ; *rs_in0 + b0 ; relu ----
  gemm_wmma_bf16<false, 8, false><<<dim3((n0 + 63) / 64), 128, 0, stream>>>(
      features, W0, t0, rs_out0, n0, HID, IND, IND, IND);
  fill_zero_kernel<<<((n0 * HID) + 255) / 256, 256, 0, stream>>>(agg0, n0 * HID);
  edge_agg_kernel<<<((e0 * (HID >> 2)) + 255) / 256, 256, 0, stream>>>(
      t0, src0, dst0, agg0, e0, HID);
  finish_bf16_kernel<<<dim3(1, n0), 128, 0, stream>>>(agg0, rs_in0, b0, h0bf, HID);

  // ---- projection 1: h1in = P1^T @ h0 (800MB stream, pipelined, split-K) ----
  {
    int kslab = ((n0 / SPLIT + 31) / 32) * 32;
    fill_zero_kernel<<<((n1v * HID) + 255) / 256, 256, 0, stream>>>(h1in, n1v * HID);
    gemm_wmma_bf16<true, 8, true>
        <<<dim3((n1v + 63) / 64, SPLIT), 128, 0, stream>>>(
            P1, h0bf, h1in, nullptr, n1v, HID, n0, n1v, kslab);
  }

  // ---- layer 1 ----
  gemm_wmma_bf16<false, 8, false><<<dim3((n1v + 63) / 64), 128, 0, stream>>>(
      h1in, W1, t1, rs_out1, n1v, HID, HID, HID, HID);
  fill_zero_kernel<<<((n1v * HID) + 255) / 256, 256, 0, stream>>>(agg1, n1v * HID);
  edge_agg_kernel<<<((e1 * (HID >> 2)) + 255) / 256, 256, 0, stream>>>(
      t1, src1, dst1, agg1, e1, HID);
  finish_bf16_kernel<<<dim3(1, n1v), 128, 0, stream>>>(agg1, rs_in1, b1, h1bf, HID);

  // ---- projection 2: emb = P2^T @ h1 (pipelined, split-K), into d_out ----
  {
    int kslab = ((n1v / SPLIT + 31) / 32) * 32;
    fill_zero_kernel<<<((n2 * HID) + 255) / 256, 256, 0, stream>>>(emb, n2 * HID);
    gemm_wmma_bf16<true, 8, true>
        <<<dim3((n2 + 63) / 64, SPLIT), 128, 0, stream>>>(
            P2, h1bf, emb, nullptr, n2, HID, n1v, n2, kslab);
  }

  // ---- layer 2 (no relu), logits into d_out[0 : n2*NC] ----
  gemm_wmma_bf16<false, 3, false><<<dim3((n2 + 63) / 64), 128, 0, stream>>>(
      emb, W2, t2, rs_out2, n2, NC, HID, HID, HID);
  fill_zero_kernel<<<((n2 * NC) + 255) / 256, 256, 0, stream>>>(agg2, n2 * NC);
  edge_agg_kernel<<<((e2 * (NC >> 2)) + 255) / 256, 256, 0, stream>>>(
      t2, src2, dst2, agg2, e2, NC);
  finish_kernel<<<dim3(1, n2), 64, 0, stream>>>(agg2, rs_in2, b2, out, NC);
}
